// TransformerBlock_53541062312296
// MI455X (gfx1250) — matrix operand order repaired and checked by reading, not compiled, not run on hardware
//
#include <hip/hip_runtime.h>
#include <math.h>
#include <stdint.h>

// ---------------------------------------------------------------------------
// Types for CDNA5 (gfx1250) WMMA: V_WMMA_F32_16X16X32_BF16, wave32.
// ---------------------------------------------------------------------------
typedef __bf16 bf16;
typedef __attribute__((ext_vector_type(8)))  bf16  v8bf;
typedef __attribute__((ext_vector_type(16))) bf16  v16bf;
typedef __attribute__((ext_vector_type(8)))  float v8f;

static constexpr int Bc = 2;
static constexpr int Sc = 2048;
static constexpr int Ec = 1024;
static constexpr int Hc = 16;
static constexpr int Dc = 64;      // head dim
static constexpr int Mc = Bc * Sc; // 4096 token rows
static constexpr int Fc = 4 * Ec;  // 4096 MLP hidden

static __device__ __forceinline__ v16bf cat16(v8bf lo, v8bf hi) {
  return __builtin_shufflevector(lo, hi, 0, 1, 2, 3, 4, 5, 6, 7,
                                 8, 9, 10, 11, 12, 13, 14, 15);
}

static __device__ __forceinline__ v8f wmma_bf16(v16bf a, v16bf b, v8f c) {
  // (neg_a, A, neg_b, B, c_mod, C, reuse_a, reuse_b)
  return __builtin_amdgcn_wmma_f32_16x16x32_bf16(false, a, false, b,
                                                 (short)0, c, false, false);
}

// Async copy 16 B from global memory into LDS (GLOBAL_LOAD_ASYNC_TO_LDS_B128,
// tracked by ASYNCcnt).  lds_addr = workgroup-relative LDS byte address
// (low 32 bits of the generic shared pointer).
static __device__ __forceinline__ void async_g2l_b128(unsigned lds_addr,
                                                      const bf16* gaddr) {
  asm volatile("global_load_async_to_lds_b128 %0, %1, off"
               :
               : "v"(lds_addr), "v"((unsigned long long)(uintptr_t)gaddr)
               : "memory");
}

static __device__ __forceinline__ unsigned lds_off(const void* p) {
  return (unsigned)(uintptr_t)p; // generic shared ptr: low dword == LDS offset
}

static __device__ __forceinline__ float gelu_tanh(float x) {
  const float c0 = 0.7978845608028654f; // sqrt(2/pi)
  return 0.5f * x * (1.0f + tanhf(c0 * (x + 0.044715f * x * x * x)));
}

// ---------------------------------------------------------------------------
// fp32 -> bf16 elementwise convert (weights)
// ---------------------------------------------------------------------------
__global__ void cvt_bf16_kernel(const float* __restrict__ src,
                                bf16* __restrict__ dst, int n) {
  int i = blockIdx.x * blockDim.x + threadIdx.x;
  if (i < n) dst[i] = (bf16)src[i];
}

// ---------------------------------------------------------------------------
// LayerNorm (torch-style: unbiased std, eps added to std) -> bf16 rows.
// One 256-thread block (8 wave32) per E=1024 row.
// ---------------------------------------------------------------------------
__global__ __launch_bounds__(256)
void layernorm_bf16_kernel(const float* __restrict__ src,
                           const float* __restrict__ scale,
                           const float* __restrict__ shift,
                           bf16* __restrict__ dst) {
  const int row = blockIdx.x;
  const float* x = src + (size_t)row * Ec;
  float vals[4], s = 0.f, ss = 0.f;
#pragma unroll
  for (int i = 0; i < 4; ++i) {
    float v = x[threadIdx.x + 256 * i];
    vals[i] = v; s += v; ss += v * v;
  }
#pragma unroll
  for (int off = 16; off >= 1; off >>= 1) {
    s  += __shfl_xor(s,  off, 32);
    ss += __shfl_xor(ss, off, 32);
  }
  __shared__ float sh_s[8], sh_q[8];
  const int wid = threadIdx.x >> 5;
  if ((threadIdx.x & 31) == 0) { sh_s[wid] = s; sh_q[wid] = ss; }
  __syncthreads();
  if (wid == 0) {
    float a = (threadIdx.x < 8) ? sh_s[threadIdx.x] : 0.f;
    float b = (threadIdx.x < 8) ? sh_q[threadIdx.x] : 0.f;
#pragma unroll
    for (int off = 4; off >= 1; off >>= 1) {
      a += __shfl_xor(a, off, 32);
      b += __shfl_xor(b, off, 32);
    }
    if (threadIdx.x == 0) { sh_s[0] = a; sh_q[0] = b; }
  }
  __syncthreads();
  const float mean = sh_s[0] * (1.0f / (float)Ec);
  const float var  = fmaxf(sh_q[0] - (float)Ec * mean * mean, 0.f)
                     * (1.0f / (float)(Ec - 1));
  const float inv  = 1.0f / (sqrtf(var) + 1e-8f);
#pragma unroll
  for (int i = 0; i < 4; ++i) {
    int c = threadIdx.x + 256 * i;
    float y = scale[c] * ((vals[i] - mean) * inv) + shift[c];
    dst[(size_t)row * Ec + c] = (bf16)y;
  }
}

// ---------------------------------------------------------------------------
// NT GEMM: C[M,N] = A[MxK](bf16,row-major) * W[NxK](bf16,row-major)^T + bias.
// Block tile 128x64 (8 wave32, each wave one 16-row strip, A frag reused 4x).
// K-steps of 32 staged in LDS with double-buffered async global->LDS copies.
// Steady-state loop is branch-free: issue step k+1's DMA, s_wait_asynccnt 3
// (only step k's group), barrier, load all fragments, 4 WMMAs, barrier.
// Last K-step is peeled (wait 0).
// modes: 1 = store bf16, 2 = GELU -> bf16, 3 = +addsrc -> fp32
// ---------------------------------------------------------------------------
__global__ __launch_bounds__(256)
void gemm_nt_wmma_kernel(const bf16* __restrict__ A,
                         const bf16* __restrict__ W,
                         const float* __restrict__ bias,
                         int Mdim, int Ndim, int Kdim,
                         float* __restrict__ outf,
                         bf16* __restrict__ outh,
                         const float* __restrict__ addsrc,
                         int mode) {
  __shared__ bf16 sA[2][128 * 32];   // 16 KB
  __shared__ bf16 sB[2][64 * 32];    //  8 KB

  const int tid  = threadIdx.x;
  const int lane = tid & 31;
  const int wid  = tid >> 5;
  const int numNT = Ndim >> 6;
  const int m0 = (blockIdx.x / numNT) * 128;
  const int n0 = (blockIdx.x % numNT) * 64;

  // Per-thread async-copy slots.
  // A tile: 128 rows x 32 k = 4096 elems; 16 per thread (two b128).
  const int arow = tid >> 1, ahalf = (tid & 1) * 16;
  const bf16* agsrc = A + (size_t)(m0 + arow) * Kdim + ahalf;
  const unsigned alds[2] = { lds_off(&sA[0][arow * 32 + ahalf]),
                             lds_off(&sA[1][arow * 32 + ahalf]) };
  // B tile: 64 rows x 32 k = 2048 elems; 8 per thread (one b128).
  const int brow = tid >> 2, bq8 = (tid & 3) * 8;
  const bf16* bgsrc = W + (size_t)(n0 + brow) * Kdim + bq8;
  const unsigned blds[2] = { lds_off(&sB[0][brow * 32 + bq8]),
                             lds_off(&sB[1][brow * 32 + bq8]) };

  const int NS = Kdim >> 5;  // always >= 2 here

  // Prologue: stage k-step 0 into buffer 0.
  async_g2l_b128(alds[0],      agsrc);
  async_g2l_b128(alds[0] + 16, agsrc + 8);
  async_g2l_b128(blds[0],      bgsrc);

  // Fragment read addresses (from LDS).
  const int akoff = (lane >> 4) * 8;   // A-frag K chunks {0-7,16-23}/{8-15,24-31}
  const int bkoff = (lane >> 4) * 8;   // B-frag K chunks {0-7,16-23}/{8-15,24-31}
  const int fr = (wid * 16 + (lane & 15)) * 32;  // this wave's A row in tile
  v8f acc[4] = {};

  for (int ks = 0; ks < NS - 1; ++ks) {
    const int cur = ks & 1;
    const int nxt = cur ^ 1;
    const int k0n = (ks + 1) << 5;
    async_g2l_b128(alds[nxt],      agsrc + k0n);
    async_g2l_b128(alds[nxt] + 16, agsrc + k0n + 8);
    async_g2l_b128(blds[nxt],      bgsrc + k0n);
    asm volatile("s_wait_asynccnt 0x3" ::: "memory"); // only step ks's group
    __syncthreads();  // all threads' async writes visible

    // Load all fragments first so ds_loads overlap the WMMA chain.
    const bf16* ap = &sA[cur][fr];
    v16bf af = cat16(*(const v8bf*)(ap + akoff),
                     *(const v8bf*)(ap + 16 + akoff));
    v16bf bfr[4];
#pragma unroll
    for (int t = 0; t < 4; ++t) {
      const bf16* bp = &sB[cur][(t * 16 + (lane & 15)) * 32 + bkoff];
      bfr[t] = cat16(*(const v8bf*)(bp), *(const v8bf*)(bp + 16));
    }
#pragma unroll
    for (int t = 0; t < 4; ++t) acc[t] = wmma_bf16(af, bfr[t], acc[t]);
    __syncthreads();  // done reading buf `cur`; safe to overwrite next round
  }

  // Peeled last K-step.
  {
    const int cur = (NS - 1) & 1;
    asm volatile("s_wait_asynccnt 0x0" ::: "memory");
    __syncthreads();
    const bf16* ap = &sA[cur][fr];
    v16bf af = cat16(*(const v8bf*)(ap + akoff),
                     *(const v8bf*)(ap + 16 + akoff));
    v16bf bfr[4];
#pragma unroll
    for (int t = 0; t < 4; ++t) {
      const bf16* bp = &sB[cur][(t * 16 + (lane & 15)) * 32 + bkoff];
      bfr[t] = cat16(*(const v8bf*)(bp), *(const v8bf*)(bp + 16));
    }
#pragma unroll
    for (int t = 0; t < 4; ++t) acc[t] = wmma_bf16(af, bfr[t], acc[t]);
  }

  // C layout: VGPR r, lanes 0-15 -> (row +r, col n0+lane); lanes 16-31 -> (+r+8).
  const int col0 = lane & 15;
  const int roff = (lane >> 4) * 8;
  const int mrow = m0 + wid * 16;
#pragma unroll
  for (int t = 0; t < 4; ++t) {
    const int col = n0 + t * 16 + col0;
    const float bv = bias[col];
#pragma unroll
    for (int r = 0; r < 8; ++r) {
      const size_t idx = (size_t)(mrow + r + roff) * Ndim + col;
      float val = acc[t][r] + bv;
      if (mode == 2) {
        outh[idx] = (bf16)gelu_tanh(val);
      } else if (mode == 1) {
        outh[idx] = (bf16)val;
      } else { // mode 3: residual add, fp32 out
        outf[idx] = addsrc[idx] + val;
      }
    }
  }
}

// ---------------------------------------------------------------------------
// Per-head V transpose: v[b,s,h,d] -> vT[b,h,d,s] so P*V B-operand loads are
// K(=key)-contiguous.
// ---------------------------------------------------------------------------
__global__ void vtranspose_kernel(const bf16* __restrict__ v,
                                  bf16* __restrict__ vT) {
  int idx = blockIdx.x * blockDim.x + threadIdx.x; // over Mc*Ec
  int row = idx >> 10;      // token row = b*S + s
  int col = idx & 1023;     // h*64 + d
  int b = row >> 11;
  int s_ = row & 2047;
  int h = col >> 6;
  int d = col & 63;
  vT[(((size_t)(b * Hc + h) * Dc + d) << 11) + s_] = v[idx];
}

// ---------------------------------------------------------------------------
// Flash attention (non-causal, no 1/sqrt(d) scale -- matches reference).
// One wave per (b, h, 16-query block). Computes scores^T = K_blk * Q^T so the
// probability tile lands directly in the WMMA A-fragment layout (no shuffles).
// Epilogue fuses the residual: out_res = inputs + attn_out.
// ---------------------------------------------------------------------------
__global__ __launch_bounds__(256)
void attention_kernel(const bf16* __restrict__ q,
                      const bf16* __restrict__ k,
                      const bf16* __restrict__ vT,
                      const float* __restrict__ inputs,
                      float* __restrict__ out_res) {
  const int lane = threadIdx.x & 31;
  const int gw   = blockIdx.x * (blockDim.x >> 5) + (threadIdx.x >> 5);
  const int QB   = Sc / 16;                // 128 query blocks
  const int q0   = (gw % QB) * 16;
  const int bh   = gw / QB;
  const int h    = bh % Hc;
  const int b    = bh / Hc;

  const int hoff = (lane >> 4) * 8;        // row-half offset (queries/keys)

  // Q as B-operand (K-dim = d): lane l -> query q0+(l&15), d chunk by half.
  const size_t qrow = (size_t)(b * Sc + q0 + (lane & 15)) * Ec + h * Dc;
  const int dk = (lane >> 4) * 8;
  const v16bf bq0 = cat16(*(const v8bf*)(q + qrow + dk),
                          *(const v8bf*)(q + qrow + dk + 16));
  const v16bf bq1 = cat16(*(const v8bf*)(q + qrow + 32 + dk),
                          *(const v8bf*)(q + qrow + 32 + dk + 16));

  float m_run = -1e30f, l_run = 0.f;
  v8f acc[4] = {};
  const int akoff = (lane >> 4) * 8;       // A-fragment d chunks for K rows
  const size_t vhead = (size_t)(b * Hc + h) * Dc;

  for (int jb = 0; jb < Sc; jb += 32) {
    // scores^T tiles: sA (keys jb..jb+15), sB (keys jb+16..jb+31)
    const size_t krA = (size_t)(b * Sc + jb + (lane & 15)) * Ec + h * Dc;
    const size_t krB = krA + (size_t)16 * Ec;
    v16bf ka0 = cat16(*(const v8bf*)(k + krA + akoff),
                      *(const v8bf*)(k + krA + 16 + akoff));
    v16bf ka1 = cat16(*(const v8bf*)(k + krA + 32 + akoff),
                      *(const v8bf*)(k + krA + 48 + akoff));
    v16bf kb0 = cat16(*(const v8bf*)(k + krB + akoff),
                      *(const v8bf*)(k + krB + 16 + akoff));
    v16bf kb1 = cat16(*(const v8bf*)(k + krB + 32 + akoff),
                      *(const v8bf*)(k + krB + 48 + akoff));
    v8f sA = {}, sB = {};
    sA = wmma_bf16(ka0, bq0, sA);
    sA = wmma_bf16(ka1, bq1, sA);
    sB = wmma_bf16(kb0, bq0, sB);
    sB = wmma_bf16(kb1, bq1, sB);

    // Per-lane = per-query (lane&15); lane l and l^16 split the 32 keys.
    float vmax = -1e30f;
#pragma unroll
    for (int r = 0; r < 8; ++r) vmax = fmaxf(vmax, fmaxf(sA[r], sB[r]));
    vmax = fmaxf(vmax, __shfl_xor(vmax, 16, 32));
    const float newm = fmaxf(m_run, vmax);
    const float corr = __expf(m_run - newm);
    m_run = newm;

    v16bf pfrag;
    float rsum = 0.f;
#pragma unroll
    for (int r = 0; r < 8; ++r) {
      float pa = __expf(sA[r] - newm);
      float pb = __expf(sB[r] - newm);
      rsum += pa + pb;
      pfrag[r]     = (bf16)pa;  // keys jb + r + hoff
      pfrag[r + 8] = (bf16)pb;  // keys jb + 16 + r + hoff
    }
    rsum += __shfl_xor(rsum, 16, 32);
    l_run = l_run * corr + rsum;

    // Rescale accumulators: acc row r maps to query r + hoff; that query's
    // corr lives in lane (r + hoff).
    float corr_g[8];
#pragma unroll
    for (int r = 0; r < 8; ++r) corr_g[r] = __shfl(corr, r + hoff, 32);
#pragma unroll
    for (int t = 0; t < 4; ++t)
#pragma unroll
      for (int r = 0; r < 8; ++r) acc[t][r] *= corr_g[r];

    // acc += P(16x32) * V(32x64): V B-operand from transposed vT (key-contig).
#pragma unroll
    for (int t = 0; t < 4; ++t) {
      const size_t vr = (vhead + t * 16 + (lane & 15)) * (size_t)Sc
                        + jb + (lane >> 4) * 8;
      v16bf bv = cat16(*(const v8bf*)(vT + vr), *(const v8bf*)(vT + vr + 16));
      acc[t] = wmma_bf16(pfrag, bv, acc[t]);
    }
  }

  // Finalize: divide by l, add residual input, store fp32.
  const float linv = 1.0f / l_run;
  float linv_g[8];
#pragma unroll
  for (int r = 0; r < 8; ++r) linv_g[r] = __shfl(linv, r + hoff, 32);
#pragma unroll
  for (int t = 0; t < 4; ++t) {
    const int col = h * Dc + t * 16 + (lane & 15);
#pragma unroll
    for (int r = 0; r < 8; ++r) {
      const size_t idx = (size_t)(b * Sc + q0 + r + hoff) * Ec + col;
      out_res[idx] = inputs[idx] + acc[t][r] * linv_g[r];
    }
  }
}

// ---------------------------------------------------------------------------
// Host launch
// ---------------------------------------------------------------------------
extern "C" void kernel_launch(void* const* d_in, const int* in_sizes, int n_in,
                              void* d_out, int out_size, void* d_ws,
                              size_t ws_size, hipStream_t stream) {
  (void)in_sizes; (void)n_in; (void)out_size; (void)ws_size;
  const float* inputs = (const float*)d_in[0];
  const float* Wq = (const float*)d_in[1];  const float* bq = (const float*)d_in[2];
  const float* Wk = (const float*)d_in[3];  const float* bk = (const float*)d_in[4];
  const float* Wv = (const float*)d_in[5];  const float* bv = (const float*)d_in[6];
  const float* W1 = (const float*)d_in[7];  const float* b1 = (const float*)d_in[8];
  const float* W2 = (const float*)d_in[9];  const float* b2 = (const float*)d_in[10];
  const float* scale1 = (const float*)d_in[11]; const float* shift1 = (const float*)d_in[12];
  const float* scale2 = (const float*)d_in[13]; const float* shift2 = (const float*)d_in[14];

  char* ws = (char*)d_ws;
  const size_t MB = 1u << 20;
  bf16*  Wq_h    = (bf16*)(ws + 0 * MB);   // 2 MB
  bf16*  Wk_h    = (bf16*)(ws + 2 * MB);   // 2 MB
  bf16*  Wv_h    = (bf16*)(ws + 4 * MB);   // 2 MB
  bf16*  W1_h    = (bf16*)(ws + 6 * MB);   // 8 MB
  bf16*  W2_h    = (bf16*)(ws + 14 * MB);  // 8 MB
  bf16*  xh      = (bf16*)(ws + 22 * MB);  // 8 MB (ln1 out; reused for ln2 out)
  bf16*  qh      = (bf16*)(ws + 30 * MB);  // 8 MB
  bf16*  kh      = (bf16*)(ws + 38 * MB);  // 8 MB
  bf16*  vh      = (bf16*)(ws + 46 * MB);  // 8 MB
  bf16*  vT      = (bf16*)(ws + 54 * MB);  // 8 MB
  float* out_res = (float*)(ws + 62 * MB); // 16 MB
  bf16*  h1g     = (bf16*)(ws + 30 * MB);  // 32 MB alias over qh..vT (free then)

  const int T = 256;
  // Weight conversion fp32 -> bf16
  cvt_bf16_kernel<<<(Ec * Ec) / T, T, 0, stream>>>(Wq, Wq_h, Ec * Ec);
  cvt_bf16_kernel<<<(Ec * Ec) / T, T, 0, stream>>>(Wk, Wk_h, Ec * Ec);
  cvt_bf16_kernel<<<(Ec * Ec) / T, T, 0, stream>>>(Wv, Wv_h, Ec * Ec);
  cvt_bf16_kernel<<<(Fc * Ec) / T, T, 0, stream>>>(W1, W1_h, Fc * Ec);
  cvt_bf16_kernel<<<(Ec * Fc) / T, T, 0, stream>>>(W2, W2_h, Ec * Fc);

  // LN1
  layernorm_bf16_kernel<<<Mc, T, 0, stream>>>(inputs, scale1, shift1, xh);

  // QKV projections (bf16 out): grid = (M/128) * (N/64) blocks
  const int gQKV = (Mc / 128) * (Ec / 64); // 512 blocks
  gemm_nt_wmma_kernel<<<gQKV, T, 0, stream>>>(xh, Wq_h, bq, Mc, Ec, Ec,
                                              nullptr, qh, nullptr, 1);
  gemm_nt_wmma_kernel<<<gQKV, T, 0, stream>>>(xh, Wk_h, bk, Mc, Ec, Ec,
                                              nullptr, kh, nullptr, 1);
  gemm_nt_wmma_kernel<<<gQKV, T, 0, stream>>>(xh, Wv_h, bv, Mc, Ec, Ec,
                                              nullptr, vh, nullptr, 1);

  // V -> per-head transposed layout
  vtranspose_kernel<<<(Mc * Ec) / T, T, 0, stream>>>(vh, vT);

  // Flash attention + residual
  const int gATT = (Bc * Hc * (Sc / 16)) / 8; // 512 blocks
  attention_kernel<<<gATT, T, 0, stream>>>(qh, kh, vT, inputs, out_res);

  // LN2 (reuse xh)
  layernorm_bf16_kernel<<<Mc, T, 0, stream>>>(out_res, scale2, shift2, xh);

  // MLP: GELU(x @ W1^T + b1) @ W2^T + b2 + out_res -> d_out
  const int gMLP1 = (Mc / 128) * (Fc / 64); // 2048 blocks
  gemm_nt_wmma_kernel<<<gMLP1, T, 0, stream>>>(xh, W1_h, b1, Mc, Fc, Ec,
                                               nullptr, h1g, nullptr, 2);
  const int gMLP2 = (Mc / 128) * (Ec / 64); // 512 blocks
  gemm_nt_wmma_kernel<<<gMLP2, T, 0, stream>>>(h1g, W2_h, b2, Mc, Ec, Fc,
                                               (float*)d_out, nullptr, out_res, 3);
}
